// LearnedTaskSpecificLinear_24910810316925
// MI455X (gfx1250) — matrix operand order repaired and checked by reading, not compiled, not run on hardware
//
#include <hip/hip_runtime.h>

// LearnedTaskSpecificLinear: out[n,:] = x[n,:] @ W[task_ids[n]]
// ROWS=2048, K=N=512, TASKS=64. f32 data, f16 WMMA with f32 accumulate.

typedef __attribute__((ext_vector_type(16))) _Float16 v16h;
typedef __attribute__((ext_vector_type(8)))  _Float16 v8h;
typedef __attribute__((ext_vector_type(4)))  _Float16 v4h;
typedef __attribute__((ext_vector_type(8)))  float    v8f;

constexpr int ROWS  = 2048;
constexpr int KDIM  = 512;   // reduction dim
constexpr int NDIM  = 512;   // output dim
constexpr int TASKS = 64;
constexpr int NB    = 128;   // output cols per block (8 waves x 16)
constexpr int KT    = 32;    // K per WMMA (f16 16x16x32)
constexpr int NSLAB = KDIM / KT;      // 16 K-slabs
constexpr int MT    = 32;    // row tile: 2 WMMA M-tiles sharing each B fragment
constexpr int AP    = 8;     // s_a pad (f16) -> row stride 260 dwords == 4 mod 64 banks
constexpr int LNS   = 24;    // s_b per-lane-slot stride (16 data f16 + 8 pad)
constexpr int WVS   = 32 * LNS + 8;   // s_b per-wave stride = 776 f16 (388 dw == 4 mod 64)

__global__ __launch_bounds__(256)
void task_linear_wmma(const float* __restrict__ x,
                      const int*   __restrict__ ids,
                      const float* __restrict__ W,
                      float*       __restrict__ out)
{
    __shared__ unsigned short s_rows[ROWS];
    __shared__ int s_cnt;
    // A: 32 gathered rows, full K, f16, padded stride (2-way worst bank conflict)
    __shared__ __attribute__((aligned(32))) _Float16 s_a[MT][KDIM + AP];
    // B: double-buffered, stored directly in WMMA B-fragment order:
    //    [buf][wave*WVS + ln*LNS + e], e = (k&7) | ((k&16)>>1), ln = ((k>>3)&1)*16 | (n&15)
    __shared__ __attribute__((aligned(32))) _Float16 s_b[2][8 * WVS];

    const int tid  = threadIdx.x;
    const int lane = tid & 31;          // wave32
    const int wv   = tid >> 5;          // 8 waves, wave wv owns cols [wv*16, wv*16+16)
    const int t    = blockIdx.y;        // task
    const int c0   = blockIdx.x * NB;   // column base

    // ---- row list for this task (value-deterministic regardless of atomic order) ----
    if (tid == 0) s_cnt = 0;
    __syncthreads();
    for (int i = tid; i < ROWS; i += 256)
        if (ids[i] == t) s_rows[atomicAdd(&s_cnt, 1)] = (unsigned short)i;
    __syncthreads();
    const int cnt = s_cnt;

    const float* Wt = W + (size_t)t * KDIM * NDIM + c0;

    // W slab stager: 32k x 128n f32 -> f16 fragment order, one unit per thread.
    // Thread loads 4 consecutive k-rows (coalesced float4) and packs 4 f16 along e.
    auto stage_w = [&](int kk, int buf) {
        const int k0 = (tid >> 5) * 4;        // 0,4,...,28
        const int n0 = (tid & 31) * 4;        // 0..124
        const float* p = Wt + (size_t)(kk + k0) * NDIM + n0;
        float4 r0 = *(const float4*)(p);
        float4 r1 = *(const float4*)(p + NDIM);
        float4 r2 = *(const float4*)(p + 2 * NDIM);
        float4 r3 = *(const float4*)(p + 3 * NDIM);
        float rr[4][4] = { {r0.x, r0.y, r0.z, r0.w}, {r1.x, r1.y, r1.z, r1.w},
                           {r2.x, r2.y, r2.z, r2.w}, {r3.x, r3.y, r3.z, r3.w} };
        const int lgrp16 = (k0 & 8) << 1;
        const int e0     = (k0 & 7) | ((k0 & 16) >> 1);   // 0,4,8,12 -> 8B-aligned b64 stores
        #pragma unroll
        for (int j = 0; j < 4; ++j) {
            int n  = n0 + j;
            int w  = n >> 4;
            int ln = lgrp16 | (n & 15);
            v4h pk = { (_Float16)rr[0][j], (_Float16)rr[1][j],
                       (_Float16)rr[2][j], (_Float16)rr[3][j] };
            *(v4h*)&s_b[buf][w * WVS + ln * LNS + e0] = pk;   // conflict-free b64
        }
    };

    // A-fragment addressing (ISA 7.12.2, 16-bit A 16x32):
    // lane L: M = L&15, kb = (L>>4)*8 ; elems 0..7 <- K kb..kb+7, 8..15 <- kb+16..kb+23
    const int m_ = lane & 15;
    const int kb = (lane >> 4) * 8;

    for (int r0 = 0; r0 < cnt; r0 += MT) {
        // ---- stage 32 gathered x rows -> f16 LDS (coalesced float4) ----
        for (int q = tid; q < MT * (KDIM / 4); q += 256) {
            int rl = q >> 7;                  // 128 float4 per row
            int c4 = q & 127;
            float4 v = make_float4(0.f, 0.f, 0.f, 0.f);
            if (r0 + rl < cnt)
                v = *(const float4*)(x + (size_t)s_rows[r0 + rl] * KDIM + c4 * 4);
            v4h h = { (_Float16)v.x, (_Float16)v.y, (_Float16)v.z, (_Float16)v.w };
            *(v4h*)&s_a[rl][c4 * 4] = h;
        }
        stage_w(0, 0);                        // prologue: slab 0 -> buffer 0
        __syncthreads();

        v8f acc0 = {};
        v8f acc1 = {};
        for (int s = 0; s < NSLAB; ++s) {
            const int kk = s * KT;
            const int p  = s & 1;
            if (s + 1 < NSLAB) stage_w(kk + KT, p ^ 1);   // overlap next slab

            // A fragments for both 16-row sub-tiles (B fragment shared)
            v8h a0lo = *(const v8h*)&s_a[m_][kk + kb];
            v8h a0hi = *(const v8h*)&s_a[m_][kk + kb + 16];
            v8h a1lo = *(const v8h*)&s_a[m_ + 16][kk + kb];
            v8h a1hi = *(const v8h*)&s_a[m_ + 16][kk + kb + 16];
            v16h a0 = __builtin_shufflevector(a0lo, a0hi,
                         0,1,2,3,4,5,6,7,8,9,10,11,12,13,14,15);
            v16h a1 = __builtin_shufflevector(a1lo, a1hi,
                         0,1,2,3,4,5,6,7,8,9,10,11,12,13,14,15);

            const _Float16* bp = &s_b[p][wv * WVS + lane * LNS];
            v8h blo = *(const v8h*)(bp);      // e 0..7   (16B aligned)
            v8h bhi = *(const v8h*)(bp + 8);  // e 8..15  (16B aligned)
            v16h b  = __builtin_shufflevector(blo, bhi,
                         0,1,2,3,4,5,6,7,8,9,10,11,12,13,14,15);

            acc0 = __builtin_amdgcn_wmma_f32_16x16x32_f16(
                       false, a0, false, b, (short)0, acc0, false, false);
            acc1 = __builtin_amdgcn_wmma_f32_16x16x32_f16(
                       false, a1, false, b, (short)0, acc1, false, false);
            __syncthreads();                  // one barrier per slab (ping-pong)
        }

        // ---- scatter results (C/D layout: VGPR r -> M = r + 8*(lane>=16)) ----
        const int col = c0 + (wv << 4) + (lane & 15);
        #pragma unroll
        for (int r = 0; r < 8; ++r) {
            int m = r + ((lane >> 4) << 3);
            if (r0 + m < cnt)
                out[(size_t)s_rows[r0 + m] * NDIM + col] = acc0[r];
            if (r0 + 16 + m < cnt)
                out[(size_t)s_rows[r0 + 16 + m] * NDIM + col] = acc1[r];
        }
        __syncthreads();                      // protect s_a / s_b before next tile
    }
}

extern "C" void kernel_launch(void* const* d_in, const int* in_sizes, int n_in,
                              void* d_out, int out_size, void* d_ws, size_t ws_size,
                              hipStream_t stream) {
    const float* x   = (const float*)d_in[0];   // [2048, 512] f32
    const int*   ids = (const int*)  d_in[1];   // [2048] int
    const float* W   = (const float*)d_in[2];   // [64, 512, 512] f32
    float*       out = (float*)d_out;           // [2048, 512] f32

    dim3 grid(NDIM / NB, TASKS);                // 4 column tiles x 64 tasks
    task_linear_wmma<<<grid, 256, 0, stream>>>(x, ids, W, out);
}
